// Node_Edge_64166811403051
// MI455X (gfx1250) — matrix operand order repaired and checked by reading, not compiled, not run on hardware
//
#include <hip/hip_runtime.h>
#include <hip/hip_bf16.h>

typedef __attribute__((ext_vector_type(16))) _Float16 v16h;
typedef __attribute__((ext_vector_type(8)))  _Float16 v8h;
typedef __attribute__((ext_vector_type(8)))  float    v8f;

#define NN   18
#define C1   64
#define C2   128
#define NPIX (NN * NN)      // 324
#define NTILES 21           // ceil(324/16)
#define H1STRIDE 136        // 128 + 8 halfs pad (bank-conflict avoidance)

// ---------------- pre-kernel: convert weights f32 -> f16 in d_ws ----------------
__global__ __launch_bounds__(256) void cvt_weights(const float* __restrict__ W1,
                                                   const float* __restrict__ W2,
                                                   _Float16* __restrict__ w1h,
                                                   _Float16* __restrict__ w2h) {
    int i = blockIdx.x * 256 + threadIdx.x;   // 0..8191
    if (i < C2 * C1) w1h[i] = (_Float16)W1[i];
    if (i < C1 * C2) w2h[i] = (_Float16)W2[i];
}

// ---------------- main kernel: one workgroup per batch ----------------
__global__ __launch_bounds__(256) void edge_kernel(
    const float* __restrict__ node,       // [B,18,64]
    const float* __restrict__ last_edge,  // [B,18,18]
    const _Float16* __restrict__ w1h,     // [128,64] f16
    const float* __restrict__ g1, const float* __restrict__ b1,
    const float* __restrict__ m1, const float* __restrict__ v1,
    const _Float16* __restrict__ w2h,     // [64,128] f16
    const float* __restrict__ g2, const float* __restrict__ b2,
    const float* __restrict__ m2, const float* __restrict__ v2,
    const float* __restrict__ W3,         // [64]
    const float* __restrict__ b3,         // [1]
    float* __restrict__ out)              // [B,18,18,2]
{
    __shared__ float    nodeS[NN * C1];            // 4608 B
    __shared__ _Float16 h1S[8 * 16 * H1STRIDE];    // per-wave 16x128 (padded) f16
    __shared__ float    sc1[C2], bi1[C2];
    __shared__ float    sc2[C1], bi2[C1];
    __shared__ float    w3S[C1];
    __shared__ float    epS[NTILES * 16];          // 336 slots

    const int b    = blockIdx.x;
    const int tid  = threadIdx.x;
    const int lane = tid & 31;
    const int wave = tid >> 5;
    const int hi   = lane >> 4;    // lane half (K sub-block selector)
    const int ln   = lane & 15;    // row (A) / column (B,D) index

    // ---------------- stage batch data + folded BN params ----------------
    for (int idx = tid; idx < NN * C1; idx += 256)
        nodeS[idx] = node[(size_t)b * NN * C1 + idx];
    if (tid < C2) {
        float s = g1[tid] * rsqrtf(v1[tid] + 1e-5f);
        sc1[tid] = s;
        bi1[tid] = b1[tid] - m1[tid] * s;
    }
    if (tid < C1) {
        float s = g2[tid] * rsqrtf(v2[tid] + 1e-5f);
        sc2[tid] = s;
        bi2[tid] = b2[tid] - m2[tid] * s;
        w3S[tid] = W3[tid];
    }
    __syncthreads();

    const float bias3 = b3[0];
    _Float16* myH1 = &h1S[wave * 16 * H1STRIDE];

    for (int tile = wave; tile < NTILES; tile += 8) {
        // This lane's A-matrix row (pixel) in the 16-row tile
        int p = tile * 16 + ln;
        if (p >= NPIX) p = NPIX - 1;            // clamp padding rows (not stored)
        const int pi = p / NN, pj = p % NN;
        const float* ni = &nodeS[pi * C1];
        const float* nj = &nodeS[pj * C1];

        // ---- A fragments for GEMM1: sim[p, c] = (node_i[c]-node_j[c])^2 ----
        // A(16x32) f16 layout: lane half h holds K = kb*32 + {h*8+0..7, 16+h*8+0..7}
        v16h a1[2];
        #pragma unroll
        for (int kb = 0; kb < 2; ++kb) {
            #pragma unroll
            for (int t = 0; t < 2; ++t) {
                const int cb = kb * 32 + t * 16 + hi * 8;
                #pragma unroll
                for (int k = 0; k < 8; ++k) {
                    float d = ni[cb + k] - nj[cb + k];
                    a1[kb][t * 8 + k] = (_Float16)(d * d);
                }
            }
        }

        // ---- GEMM1 (K=64 -> 128 ch) + BN1 + LeakyReLU -> h1 tile in LDS ----
        #pragma unroll
        for (int nt = 0; nt < 8; ++nt) {
            v8f acc = {};
            #pragma unroll
            for (int kb = 0; kb < 2; ++kb) {
                // B(32x16) layout: lane half h holds K = kb*32 + {h*8+0..7, 16+h*8+0..7}, col o
                const v8h* wp = (const v8h*)&w1h[(nt * 16 + ln) * C1 + kb * 32 + hi * 8];
                union { v16h v; v8h h[2]; } bu;
                bu.h[0] = wp[0];
                bu.h[1] = wp[2];
                acc = __builtin_amdgcn_wmma_f32_16x16x32_f16(
                    false, a1[kb], false, bu.v, (short)0, acc, false, false);
            }
            const int o = nt * 16 + ln;          // D col = output channel
            const float s = sc1[o], bb = bi1[o];
            #pragma unroll
            for (int v = 0; v < 8; ++v) {        // D row = v + 8*hi (pixel in tile)
                float h = acc[v] * s + bb;
                h = (h >= 0.f) ? h : 0.01f * h;
                myH1[(v + 8 * hi) * H1STRIDE + o] = (_Float16)h;
            }
        }
        // DS ops within a wave complete in-order; just stop compiler reordering.
        asm volatile("" ::: "memory");

        // ---- Reload h1 in A layout (K=128 -> 4 fragments) ----
        v16h a2[4];
        #pragma unroll
        for (int kb = 0; kb < 4; ++kb) {
            union { v16h v; v8h h[2]; } au;
            au.h[0] = *(const v8h*)&myH1[ln * H1STRIDE + kb * 32 + hi * 8];
            au.h[1] = *(const v8h*)&myH1[ln * H1STRIDE + kb * 32 + 16 + hi * 8];
            a2[kb] = au.v;
        }

        // ---- GEMM2 (K=128 -> 64 ch) + BN2 + LeakyReLU + dot(W3), fused ----
        v8f part = {};
        #pragma unroll
        for (int nt = 0; nt < 4; ++nt) {
            v8f acc = {};
            #pragma unroll
            for (int kb = 0; kb < 4; ++kb) {
                const v8h* wp = (const v8h*)&w2h[(nt * 16 + ln) * C2 + kb * 32 + hi * 8];
                union { v16h v; v8h h[2]; } bu;
                bu.h[0] = wp[0];
                bu.h[1] = wp[2];
                acc = __builtin_amdgcn_wmma_f32_16x16x32_f16(
                    false, a2[kb], false, bu.v, (short)0, acc, false, false);
            }
            const int o = nt * 16 + ln;
            const float s = sc2[o], bb = bi2[o], w3v = w3S[o];
            #pragma unroll
            for (int v = 0; v < 8; ++v) {
                float h = acc[v] * s + bb;
                h = (h >= 0.f) ? h : 0.01f * h;
                part[v] += h * w3v;              // GEMM3 partial (channel o)
            }
        }
        // Per-pixel channel sums live across the 16 lanes of each half: reduce.
        #pragma unroll
        for (int off = 1; off < 16; off <<= 1) {
            #pragma unroll
            for (int v = 0; v < 8; ++v)
                part[v] += __shfl_xor(part[v], off, 32);
        }
        if (ln == 0) {
            #pragma unroll
            for (int v = 0; v < 8; ++v) {
                int pp = tile * 16 + v + 8 * hi;
                if (pp < NPIX) epS[pp] = part[v] + bias3;
            }
        }
    }
    __syncthreads();

    // ---------------- mask + eye + row softmax + interleave output ----------------
    const float* le = &last_edge[(size_t)b * NPIX];
    float*       ob = &out[(size_t)b * NPIX * 2];
    for (int r = tid; r < NN; r += 256) {
        float mx = -1e30f;
        #pragma unroll
        for (int j = 0; j < NN; ++j) {
            float e = epS[r * NN + j];
            int dd = r - j; if (dd < 0) dd = -dd;
            bool cross = (r < 9) != (j < 9);
            float vv = ((dd <= 1) || cross) ? e : 0.f;   // ep * MASK
            vv += (r == j) ? 1.f : 0.f;                  // + EYE
            mx = fmaxf(mx, vv);
        }
        float sum = 0.f;
        #pragma unroll
        for (int j = 0; j < NN; ++j) {
            float e = epS[r * NN + j];
            int dd = r - j; if (dd < 0) dd = -dd;
            bool cross = (r < 9) != (j < 9);
            float vv = ((dd <= 1) || cross) ? e : 0.f;
            vv += (r == j) ? 1.f : 0.f;
            sum += __expf(vv - mx);
        }
        const float inv = 1.f / sum;
        #pragma unroll
        for (int j = 0; j < NN; ++j) {
            float e = epS[r * NN + j];
            int dd = r - j; if (dd < 0) dd = -dd;
            bool cross = (r < 9) != (j < 9);
            float vv = ((dd <= 1) || cross) ? e : 0.f;
            vv += (r == j) ? 1.f : 0.f;
            float2 o2 = make_float2(le[r * NN + j], __expf(vv - mx) * inv);
            *(float2*)&ob[(r * NN + j) * 2] = o2;
        }
    }
}

extern "C" void kernel_launch(void* const* d_in, const int* in_sizes, int n_in,
                              void* d_out, int out_size, void* d_ws, size_t ws_size,
                              hipStream_t stream) {
    const float* node      = (const float*)d_in[0];
    const float* last_edge = (const float*)d_in[1];
    const float* W1        = (const float*)d_in[2];
    const float* g1        = (const float*)d_in[3];
    const float* b1        = (const float*)d_in[4];
    const float* m1        = (const float*)d_in[5];
    const float* v1        = (const float*)d_in[6];
    const float* W2        = (const float*)d_in[7];
    const float* g2        = (const float*)d_in[8];
    const float* b2        = (const float*)d_in[9];
    const float* m2        = (const float*)d_in[10];
    const float* v2        = (const float*)d_in[11];
    const float* W3        = (const float*)d_in[12];
    const float* b3        = (const float*)d_in[13];

    const int B = in_sizes[0] / (NN * C1);   // 4096

    // f16 weight copies in workspace: w1h [128*64], w2h [64*128]
    _Float16* w1h = (_Float16*)d_ws;
    _Float16* w2h = w1h + C2 * C1;

    cvt_weights<<<(C2 * C1 + 255) / 256, 256, 0, stream>>>(W1, W2, w1h, w2h);
    edge_kernel<<<B, 256, 0, stream>>>(node, last_edge,
                                       w1h, g1, b1, m1, v1,
                                       w2h, g2, b2, m2, v2,
                                       W3, b3, (float*)d_out);
}
